// Model_11819749998831
// MI455X (gfx1250) — matrix operand order repaired and checked by reading, not compiled, not run on hardware
//
#include <hip/hip_runtime.h>

typedef _Float16 h16;
typedef __attribute__((ext_vector_type(16))) _Float16 v16h;
typedef __attribute__((ext_vector_type(8)))  _Float16 v8h;
typedef __attribute__((ext_vector_type(8)))  float    v8f;

// ---------------------------------------------------------------------------
// Utility kernels
// ---------------------------------------------------------------------------
__global__ void zero_kernel(float* p, long n) {
  long i = (long)blockIdx.x * blockDim.x + threadIdx.x;
  long stride = (long)gridDim.x * blockDim.x;
  for (; i < n; i += stride) p[i] = 0.f;
}

__global__ void f32_to_f16_kernel(const float* __restrict__ x, h16* __restrict__ y, long n) {
  long i = (long)blockIdx.x * blockDim.x + threadIdx.x;
  long stride = (long)gridDim.x * blockDim.x;
  for (; i < n; i += stride) y[i] = (h16)x[i];
}

// Repack f32 weights [ntaps][Cin][Cout] into WMMA B-operand layout (f16):
// element index = (((k*nKC + c)*nNT + jt)*32 + lane)*16 + t
// lane n (0..15) holds column n, K = c*32 + {0..7,16..23} ; lane n+16 holds K = c*32 + {8..15,24..31}
__global__ void repack_kernel(const float* __restrict__ W, h16* __restrict__ Wp,
                              int ntaps, int Cin, int Cout, int nKC, int nNT) {
  long total = (long)ntaps * nKC * nNT * 512;
  long i = (long)blockIdx.x * blockDim.x + threadIdx.x;
  long stride = (long)gridDim.x * blockDim.x;
  for (; i < total; i += stride) {
    int  t    = (int)(i & 15);
    int  lane = (int)((i >> 4) & 31);
    long rest = i >> 9;
    int  jt = (int)(rest % nNT); rest /= nNT;
    int  c  = (int)(rest % nKC);
    int  k  = (int)(rest / nKC);
    int  kk  = c * 32 + (lane >> 4) * 8 + (t & 7) + (t >> 3) * 16;
    int  col = jt * 16 + (lane & 15);
    float v = (kk < Cin && col < Cout) ? W[((long)k * Cin + kk) * Cout + col] : 0.f;
    Wp[i] = (h16)v;
  }
}

// ---------------------------------------------------------------------------
// Gather-GEMM with WMMA (templated, f16 activations):
// out[n, off+j] = sum_k sum_c feat[nbr[n,k], c] * W[k][c][j]
// One wave32 per block; 16-row x COUT-col tile; A fetched once per tap/chunk
// and reused across all COUT/16 column tiles.
// ---------------------------------------------------------------------------
template<int CIN, int COUT>
__global__ __launch_bounds__(32) void gather_gemm_wmma(
    const h16* __restrict__ feat, const int* __restrict__ nbr,
    const h16* __restrict__ Wp, float* __restrict__ out,
    int outStride, int outColOff, int Nout, int ntaps)
{
  constexpr int nKC = (CIN + 31) / 32;
  constexpr int nNT = COUT / 16;
  const int lane = threadIdx.x;
  const int m  = lane & 15;
  const int hi = lane >> 4;
  const int row0 = blockIdx.x * 16;
  const int r = row0 + m;

  v8f acc[nNT];
#pragma unroll
  for (int j = 0; j < nNT; ++j)
#pragma unroll
    for (int v = 0; v < 8; ++v) acc[j][v] = 0.f;

  for (int k = 0; k < ntaps; ++k) {
    const int idx = (r < Nout) ? nbr[(long)r * ntaps + k] : -1;
    const h16* bp = Wp + ((long)k * nKC * nNT * 32 + lane) * 16;
#pragma unroll
    for (int c = 0; c < nKC; ++c) {
      const int kbase = c * 32 + hi * 8;
      v16h a;
      if constexpr (CIN % 32 == 0) {
        if (idx >= 0) {
          const h16* fp = feat + (long)idx * CIN + kbase;
          const v8h lo = *(const v8h*)(fp);
          const v8h hh = *(const v8h*)(fp + 16);
          a = __builtin_shufflevector(lo, hh, 0,1,2,3,4,5,6,7,8,9,10,11,12,13,14,15);
        } else {
#pragma unroll
          for (int t = 0; t < 16; ++t) a[t] = (h16)0.f;
        }
      } else {
        const h16* fp = feat + (long)idx * CIN;
#pragma unroll
        for (int t = 0; t < 8; ++t) {
          const int k1 = kbase + t, k2 = kbase + 16 + t;
          a[t]     = (idx >= 0 && k1 < CIN) ? fp[k1] : (h16)0.f;
          a[t + 8] = (idx >= 0 && k2 < CIN) ? fp[k2] : (h16)0.f;
        }
      }
#pragma unroll
      for (int j = 0; j < nNT; ++j) {
        const v16h b = *(const v16h*)(bp + ((long)c * nNT + j) * 512);
        acc[j] = __builtin_amdgcn_wmma_f32_16x16x32_f16(false, a, false, b,
                                                        (short)0, acc[j], false, false);
      }
    }
  }

#pragma unroll
  for (int j = 0; j < nNT; ++j) {
    const int col = outColOff + j * 16 + m;
#pragma unroll
    for (int v = 0; v < 8; ++v) {
      const int rr = row0 + v + 8 * hi;
      if (rr < Nout) out[(long)rr * outStride + col] = acc[j][v];
    }
  }
}

// ---------------------------------------------------------------------------
// Scatter-GEMM (up_conv): per tap k, T = featc(tile) @ W[k]; atomically add
// T[m,:] into out[rb[m,k], off:off+COUT] when index valid.
// ---------------------------------------------------------------------------
template<int CIN, int COUT>
__global__ __launch_bounds__(32) void scatter_gemm_wmma(
    const h16* __restrict__ featc, const int* __restrict__ rb,
    const h16* __restrict__ Wp, float* __restrict__ out,
    int outStride, int outColOff, int Nc, int ntaps)
{
  constexpr int nKC = CIN / 32;
  constexpr int nNT = COUT / 16;
  const int lane = threadIdx.x;
  const int m  = lane & 15;
  const int hi = lane >> 4;
  const int row0 = blockIdx.x * 16;
  const int r = row0 + m;
  const bool aval = (r < Nc);

  for (int k = 0; k < ntaps; ++k) {
    v8f acc[nNT];
#pragma unroll
    for (int j = 0; j < nNT; ++j)
#pragma unroll
      for (int v = 0; v < 8; ++v) acc[j][v] = 0.f;

    const h16* bp = Wp + ((long)k * nKC * nNT * 32 + lane) * 16;
#pragma unroll
    for (int c = 0; c < nKC; ++c) {
      const int kbase = c * 32 + hi * 8;
      v16h a;
      if (aval) {
        const h16* fp = featc + (long)r * CIN + kbase;
        const v8h lo = *(const v8h*)(fp);
        const v8h hh = *(const v8h*)(fp + 16);
        a = __builtin_shufflevector(lo, hh, 0,1,2,3,4,5,6,7,8,9,10,11,12,13,14,15);
      } else {
#pragma unroll
        for (int t = 0; t < 16; ++t) a[t] = (h16)0.f;
      }
#pragma unroll
      for (int j = 0; j < nNT; ++j) {
        const v16h b = *(const v16h*)(bp + ((long)c * nNT + j) * 512);
        acc[j] = __builtin_amdgcn_wmma_f32_16x16x32_f16(false, a, false, b,
                                                        (short)0, acc[j], false, false);
      }
    }

#pragma unroll
    for (int v = 0; v < 8; ++v) {
      const int rr = row0 + v + 8 * hi;
      if (rr < Nc) {
        const int idx = rb[(long)rr * ntaps + k];
        if (idx >= 0) {
          float* op = out + (long)idx * outStride + outColOff + m;
#pragma unroll
          for (int j = 0; j < nNT; ++j) atomicAdd(op + j * 16, acc[j][v]);
        }
      }
    }
  }
}

// ---------------------------------------------------------------------------
// BatchNorm statistics + apply (apply writes f16 activations)
// ---------------------------------------------------------------------------
__global__ void colsum_kernel(const float* __restrict__ x, float* __restrict__ stats,
                              int N, int C) {
  const int c = blockIdx.x;
  float s = 0.f, s2 = 0.f;
  for (long r = (long)blockIdx.y * blockDim.x + threadIdx.x; r < N;
       r += (long)gridDim.y * blockDim.x) {
    float v = x[r * C + c];
    s += v; s2 += v * v;
  }
  __shared__ float sh[512];
  sh[threadIdx.x] = s;
  sh[256 + threadIdx.x] = s2;
  __syncthreads();
  for (int o = 128; o > 0; o >>= 1) {
    if ((int)threadIdx.x < o) {
      sh[threadIdx.x] += sh[threadIdx.x + o];
      sh[256 + threadIdx.x] += sh[256 + threadIdx.x + o];
    }
    __syncthreads();
  }
  if (threadIdx.x == 0) {
    atomicAdd(&stats[c], sh[0]);
    atomicAdd(&stats[C + c], sh[256]);
  }
}

__global__ void bn_apply_kernel(const float* __restrict__ x, const float* __restrict__ stats,
                                const float* __restrict__ gb, h16* __restrict__ y,
                                long total, int C, float invN) {
  long i = (long)blockIdx.x * blockDim.x + threadIdx.x;
  long stride = (long)gridDim.x * blockDim.x;
  for (; i < total; i += stride) {
    int c = (int)(i % C);
    float mu  = stats[c] * invN;
    float var = stats[C + c] * invN - mu * mu;
    float v = (x[i] - mu) * gb[c] * rsqrtf(var + 1e-4f) + gb[C + c];
    y[i] = (h16)(v > 0.f ? v : 0.f);
  }
}

__global__ void copy_cols_kernel(const float* __restrict__ x, float* __restrict__ dst,
                                 long N, int C, int dstStride, int off) {
  long total = N * C;
  long i = (long)blockIdx.x * blockDim.x + threadIdx.x;
  long stride = (long)gridDim.x * blockDim.x;
  for (; i < total; i += stride) {
    long r = i / C; int c = (int)(i % C);
    dst[r * dstStride + off + c] = x[i];
  }
}

__global__ void final_linear_kernel(const h16* __restrict__ x, const float* __restrict__ lw,
                                    const float* __restrict__ lb, float* __restrict__ out,
                                    int N) {
  int r = blockIdx.x * blockDim.x + threadIdx.x;
  if (r >= N) return;
  float a0 = lb[0], a1 = lb[1];
  const h16* xr = x + (long)r * 32;
#pragma unroll
  for (int c = 0; c < 32; ++c) {
    float v = (float)xr[c];
    a0 += v * lw[c * 2];
    a1 += v * lw[c * 2 + 1];
  }
  out[(long)r * 2]     = a0;
  out[(long)r * 2 + 1] = a1;
}

// ---------------------------------------------------------------------------
// Host-side helpers
// ---------------------------------------------------------------------------
static inline long lmin(long a, long b) { return a < b ? a : b; }

static void run_repack(const float* W, h16* dst, int ntaps, int Cin, int Cout,
                       hipStream_t s) {
  int nKC = (Cin + 31) / 32, nNT = Cout / 16;
  long total = (long)ntaps * nKC * nNT * 512;
  int blocks = (int)lmin((total + 255) / 256, 2048);
  repack_kernel<<<blocks, 256, 0, s>>>(W, dst, ntaps, Cin, Cout, nKC, nNT);
}

static void run_gather_conv(const h16* feat, const int* nbr, const h16* Wp, float* out,
                            int outStride, int outColOff, int Nout, int Cin, int Cout,
                            int ntaps, hipStream_t s) {
  dim3 g((Nout + 15) / 16);
#define GG(CI, CO) gather_gemm_wmma<CI, CO><<<g, 32, 0, s>>>( \
    feat, nbr, Wp, out, outStride, outColOff, Nout, ntaps)
  if      (Cin == 3   && Cout == 32) GG(3, 32);
  else if (Cin == 32  && Cout == 32) GG(32, 32);
  else if (Cin == 32  && Cout == 64) GG(32, 64);
  else if (Cin == 64  && Cout == 64) GG(64, 64);
  else if (Cin == 64  && Cout == 96) GG(64, 96);
  else if (Cin == 96  && Cout == 96) GG(96, 96);
  else if (Cin == 128 && Cout == 64) GG(128, 64);
  else if (Cin == 64  && Cout == 32) GG(64, 32);
#undef GG
}

static void run_scatter_conv(const h16* featc, const int* rb, const h16* Wp, float* out,
                             int outStride, int outColOff, int Nc, int Cin, int Cout,
                             int ntaps, hipStream_t s) {
  dim3 g((Nc + 15) / 16);
#define SG(CI, CO) scatter_gemm_wmma<CI, CO><<<g, 32, 0, s>>>( \
    featc, rb, Wp, out, outStride, outColOff, Nc, ntaps)
  if      (Cin == 96 && Cout == 64) SG(96, 64);
  else if (Cin == 64 && Cout == 32) SG(64, 32);
#undef SG
}

static void run_bn(const float* x, const float* gb, h16* y, int N, int C,
                   float* stats, hipStream_t s) {
  zero_kernel<<<1, 256, 0, s>>>(stats, 2L * C);
  dim3 g(C, 64);
  colsum_kernel<<<g, 256, 0, s>>>(x, stats, N, C);
  long total = (long)N * C;
  int blocks = (int)lmin((total + 255) / 256, 4096);
  bn_apply_kernel<<<blocks, 256, 0, s>>>(x, stats, gb, y, total, C, 1.f / (float)N);
}

static void run_zero(float* p, long n, hipStream_t s) {
  int blocks = (int)lmin((n + 255) / 256, 4096);
  zero_kernel<<<blocks, 256, 0, s>>>(p, n);
}

static void run_copy_cols(const float* x, float* dst, long N, int C, int dstStride,
                          int off, hipStream_t s) {
  long t = N * C;
  int blocks = (int)lmin((t + 255) / 256, 4096);
  copy_cols_kernel<<<blocks, 256, 0, s>>>(x, dst, N, C, dstStride, off);
}

// ---------------------------------------------------------------------------
// Entry point
// ---------------------------------------------------------------------------
extern "C" void kernel_launch(void* const* d_in, const int* in_sizes, int n_in,
                              void* d_out, int out_size, void* d_ws, size_t ws_size,
                              hipStream_t stream) {
  (void)n_in; (void)out_size; (void)ws_size;
  const float* feat     = (const float*)d_in[0];
  const float* w_in     = (const float*)d_in[1];
  const float* bn1_0    = (const float*)d_in[2];
  const float* w_sm1_0  = (const float*)d_in[3];
  const float* bnd_0    = (const float*)d_in[4];
  const float* w_down_0 = (const float*)d_in[5];
  const float* bn1_1    = (const float*)d_in[6];
  const float* w_sm1_1  = (const float*)d_in[7];
  const float* bnd_1    = (const float*)d_in[8];
  const float* w_down_1 = (const float*)d_in[9];
  const float* bn1_2    = (const float*)d_in[10];
  const float* w_sm1_2  = (const float*)d_in[11];
  const float* bnu_1    = (const float*)d_in[12];
  const float* w_up_1   = (const float*)d_in[13];
  const float* bn2_1    = (const float*)d_in[14];
  const float* w_sm2_1  = (const float*)d_in[15];
  const float* bnu_0    = (const float*)d_in[16];
  const float* w_up_0   = (const float*)d_in[17];
  const float* bn2_0    = (const float*)d_in[18];
  const float* w_sm2_0  = (const float*)d_in[19];
  const float* bnf      = (const float*)d_in[20];
  const float* lin_w    = (const float*)d_in[21];
  const float* lin_b    = (const float*)d_in[22];
  const int*   nbr_0    = (const int*)d_in[23];
  const int*   nbr_1    = (const int*)d_in[24];
  const int*   nbr_2    = (const int*)d_in[25];
  const int*   rb_0     = (const int*)d_in[26];
  const int*   rb_1     = (const int*)d_in[27];
  float* out = (float*)d_out;

  const int N0 = in_sizes[0] / 3;
  const int N1 = in_sizes[24] / 27;
  const int N2 = in_sizes[25] / 27;

  // Bump allocator over d_ws (64B-aligned chunks)
  char* ws = (char*)d_ws;
  size_t off = 0;
  auto alloc = [&](size_t bytes) {
    void* p = ws + off;
    off += (bytes + 63) & ~(size_t)63;
    return p;
  };

  // f32 intermediates
  float* region0 = (float*)alloc((size_t)N0 * 64 * 4); // t0 (first half) / cat0
  float* A0 = region0;                                 // N0 x 32 (t0)
  float* C0 = (float*)alloc((size_t)N0 * 32 * 4);      // x0 / y0
  float* D1 = (float*)alloc((size_t)N1 * 64 * 4);      // d0
  float* F1 = (float*)alloc((size_t)N1 * 64 * 4);      // x1 / y1
  float* K1 = (float*)alloc((size_t)N1 * 128 * 4);     // cat1
  float* G2 = (float*)alloc((size_t)N2 * 96 * 4);      // d1 / x2
  float* stats = (float*)alloc(1024);

  // f16 activations (BN outputs + converted input features)
  h16* hf = (h16*)alloc((size_t)N0 * 3 * 2);
  h16* h0 = (h16*)alloc((size_t)N0 * 32 * 2);
  h16* h1 = (h16*)alloc((size_t)N1 * 64 * 2);
  h16* h2 = (h16*)alloc((size_t)N2 * 96 * 2);
  h16* h3 = (h16*)alloc((size_t)N1 * 128 * 2);
  h16* h4 = (h16*)alloc((size_t)N0 * 64 * 2);

  // Packed f16 weights
  h16* wp = (h16*)alloc(0);
  auto palloc = [&](int ntaps, int Cin, int Cout) {
    size_t n = (size_t)ntaps * ((Cin + 31) / 32) * (Cout / 16) * 512;
    return (h16*)alloc(n * 2);
  };
  (void)wp;
  h16* P_in   = palloc(27, 3, 32);    run_repack(w_in,     P_in,   27, 3, 32, stream);
  h16* P_sm10 = palloc(27, 32, 32);   run_repack(w_sm1_0,  P_sm10, 27, 32, 32, stream);
  h16* P_dn0  = palloc(8, 32, 64);    run_repack(w_down_0, P_dn0,  8, 32, 64, stream);
  h16* P_sm11 = palloc(27, 64, 64);   run_repack(w_sm1_1,  P_sm11, 27, 64, 64, stream);
  h16* P_dn1  = palloc(8, 64, 96);    run_repack(w_down_1, P_dn1,  8, 64, 96, stream);
  h16* P_sm12 = palloc(27, 96, 96);   run_repack(w_sm1_2,  P_sm12, 27, 96, 96, stream);
  h16* P_up1  = palloc(8, 96, 64);    run_repack(w_up_1,   P_up1,  8, 96, 64, stream);
  h16* P_sm21 = palloc(27, 128, 64);  run_repack(w_sm2_1,  P_sm21, 27, 128, 64, stream);
  h16* P_up0  = palloc(8, 64, 32);    run_repack(w_up_0,   P_up0,  8, 64, 32, stream);
  h16* P_sm20 = palloc(27, 64, 32);   run_repack(w_sm2_0,  P_sm20, 27, 64, 32, stream);

  // Convert input features to f16 once
  {
    long t = (long)N0 * 3;
    int blocks = (int)lmin((t + 255) / 256, 4096);
    f32_to_f16_kernel<<<blocks, 256, 0, stream>>>(feat, hf, t);
  }

  // Encoder, fine level
  run_gather_conv(hf, nbr_0, P_in, A0, 32, 0, N0, 3, 32, 27, stream);        // t0
  run_bn(A0, bn1_0, h0, N0, 32, stats, stream);
  run_gather_conv(h0, nbr_0, P_sm10, C0, 32, 0, N0, 32, 32, 27, stream);     // x0
  run_bn(C0, bnd_0, h0, N0, 32, stats, stream);
  run_gather_conv(h0, rb_0, P_dn0, D1, 64, 0, N1, 32, 64, 8, stream);        // d0

  // Mid level
  run_bn(D1, bn1_1, h1, N1, 64, stats, stream);
  run_gather_conv(h1, nbr_1, P_sm11, F1, 64, 0, N1, 64, 64, 27, stream);     // x1
  run_bn(F1, bnd_1, h1, N1, 64, stats, stream);
  run_gather_conv(h1, rb_1, P_dn1, G2, 96, 0, N2, 64, 96, 8, stream);        // d1

  // Coarse level
  run_bn(G2, bn1_2, h2, N2, 96, stats, stream);
  run_gather_conv(h2, nbr_2, P_sm12, G2, 96, 0, N2, 96, 96, 27, stream);     // x2
  run_bn(G2, bnu_1, h2, N2, 96, stats, stream);                              // bn(x2)

  // Decoder, mid level: cat1 = [x1 | up(x2)]
  run_zero(K1, (long)N1 * 128, stream);
  run_copy_cols(F1, K1, N1, 64, 128, 0, stream);
  run_scatter_conv(h2, rb_1, P_up1, K1, 128, 64, N2, 96, 64, 8, stream);     // u1
  run_bn(K1, bn2_1, h3, N1, 128, stats, stream);
  run_gather_conv(h3, nbr_1, P_sm21, F1, 64, 0, N1, 128, 64, 27, stream);    // y1
  run_bn(F1, bnu_0, h1, N1, 64, stats, stream);

  // Decoder, fine level: cat0 = [x0 | up(y1)] (overlays t0 region)
  run_zero(region0, (long)N0 * 64, stream);
  run_copy_cols(C0, region0, N0, 32, 64, 0, stream);
  run_scatter_conv(h1, rb_0, P_up0, region0, 64, 32, N1, 64, 32, 8, stream); // u0
  run_bn(region0, bn2_0, h4, N0, 64, stats, stream);
  run_gather_conv(h4, nbr_0, P_sm20, C0, 32, 0, N0, 64, 32, 27, stream);     // y0
  run_bn(C0, bnf, h0, N0, 32, stats, stream);

  // Final linear 32 -> 2
  final_linear_kernel<<<(N0 + 255) / 256, 256, 0, stream>>>(h0, lin_w, lin_b, out, N0);
}
